// get_model_47648367181924
// MI455X (gfx1250) — compile-verified, matrix operand order checked by reading, not run on hardware
//
#include <hip/hip_runtime.h>
#include <math.h>

typedef __attribute__((ext_vector_type(16))) _Float16 v16h;
typedef __attribute__((ext_vector_type(8)))  _Float16 v8h;
typedef __attribute__((ext_vector_type(8)))  float    v8f;

#define BATCH 16
#define NPTS  4096
#define BN_RS rsqrtf(1.0f + 1e-5f)

// ---------------------------------------------------------------------------
// Weight prep: fold BN scale into f16 weights, pad K to KP (multiple of 32).
// Wh[n*KP+k] = (k<K) ? W[n*K+k]*g[n]/sqrt(1+eps) : 0 ;  bias2[n] = b[n]*a+bt[n]
// ---------------------------------------------------------------------------
__global__ void prep_weights_kernel(const float* W, const float* bb,
                                    const float* gg, const float* tt,
                                    int K, int KP, int NOUT,
                                    _Float16* Wh, float* bias2)
{
    int t = blockIdx.x * blockDim.x + threadIdx.x;
    if (t >= NOUT * KP) return;
    int n = t / KP, k = t % KP;
    float a = gg[n] * BN_RS;
    Wh[t] = (k < K) ? (_Float16)(W[(size_t)n * K + k] * a) : (_Float16)0.f;
    if (k == 0) bias2[n] = bb[n] * a + tt[n];
}

// ---------------------------------------------------------------------------
// WMMA micro-layer on LDS-resident f16 activations:
// Y(S x COUT) = relu( X(S x KS) * Wh^T + bias2 ), Wh f16 pre-folded, stride KS.
// ---------------------------------------------------------------------------
template<int S, int KS, int COUT>
__device__ inline void layer_lds(const _Float16* Xl, _Float16* Yl,
                                 const _Float16* Wh, const float* bias2)
{
    const int tid  = threadIdx.x;
    const int wave = tid >> 5;
    const int lane = tid & 31;
    const int gq   = lane >> 4;
    const int ln   = lane & 15;
    constexpr int NT = COUT / 16;
    constexpr int MT = S / 16;

    for (int t = wave; t < MT * NT; t += 4) {
        const int tm = t / NT, tn = t % NT;
        const int n  = tn * 16 + ln;
        const float b2 = bias2[n];
        const _Float16* wr = Wh + (size_t)n * KS;
        v8f acc = {0.f,0.f,0.f,0.f,0.f,0.f,0.f,0.f};

        for (int k0 = 0; k0 < KS; k0 += 32) {
            const _Float16* xr = Xl + (tm * 16 + ln) * KS + k0 + 8 * gq;
            v8h alo = *(const v8h*)(xr);
            v8h ahi = *(const v8h*)(xr + 16);
            v8h blo = *(const v8h*)(wr + k0 + 8 * gq);
            v8h bhi = *(const v8h*)(wr + k0 + 16 + 8 * gq);
            v16h a, bf;
#pragma unroll
            for (int i = 0; i < 8; i++) {
                a[i] = alo[i];  a[i + 8]  = ahi[i];
                bf[i] = blo[i]; bf[i + 8] = bhi[i];
            }
            acc = __builtin_amdgcn_wmma_f32_16x16x32_f16(
                false, a, false, bf, (short)0, acc, false, false);
        }
#pragma unroll
        for (int r = 0; r < 8; r++) {
            int row = tm * 16 + r + 8 * gq;
            Yl[row * COUT + n] = (_Float16)fmaxf(acc[r] + b2, 0.f);
        }
    }
}

// ---------------------------------------------------------------------------
// Fused Set-Abstraction: one workgroup = one group (b, j):
// gather+center -> 3x (WMMA GEMM + BN + relu) in LDS -> maxpool over S.
// MODE 1: features = gathered xyz (SA1). MODE 2: features = f16 pts (SA2).
// ---------------------------------------------------------------------------
template<int S, int CIN, int KP, int C1, int C2, int C3, int MODE>
__global__ __launch_bounds__(128) void sa_fused_kernel(
    const float* xyz, const _Float16* pts, const float* new_xyz, const int* gidx,
    const _Float16* W1, const float* s1,
    const _Float16* W2, const float* s2,
    const _Float16* W3, const float* s3,
    _Float16* outp, int Np, int M)
{
    constexpr int SZA = S * ((KP > C2) ? KP : C2);
    constexpr int SZB = S * ((C1 > C3) ? C1 : C3);
    __shared__ __align__(16) _Float16 BUFA[SZA];
    __shared__ __align__(16) _Float16 BUFB[SZB];

    const int blk = blockIdx.x;
    const int b   = blk / M;
    const int tid = threadIdx.x;
    const int* gi = gidx + (size_t)blk * S;
    const float cx = new_xyz[blk * 3 + 0];
    const float cy = new_xyz[blk * 3 + 1];
    const float cz = new_xyz[blk * 3 + 2];

    for (int e = tid; e < S * KP; e += 128) {
        int s = e / KP, c = e % KP;
        float v = 0.f;
        if (c < CIN) {
            int gix = gi[s];
            const float* p = xyz + ((size_t)b * Np + gix) * 3;
            if      (c == 0) v = p[0] - cx;
            else if (c == 1) v = p[1] - cy;
            else if (c == 2) v = p[2] - cz;
            else {
                if (MODE == 1) v = p[c - 3];
                else v = (float)pts[((size_t)b * Np + gix) * (CIN - 3) + (c - 3)];
            }
        }
        BUFA[e] = (_Float16)v;
    }
    __syncthreads();
    layer_lds<S, KP, C1>(BUFA, BUFB, W1, s1);
    __syncthreads();
    layer_lds<S, C1, C2>(BUFB, BUFA, W2, s2);
    __syncthreads();
    layer_lds<S, C2, C3>(BUFA, BUFB, W3, s3);
    __syncthreads();
    for (int c = tid; c < C3; c += 128) {
        float mv = -1e30f;
        for (int s = 0; s < S; s++) mv = fmaxf(mv, (float)BUFB[s * C3 + c]);
        outp[(size_t)blk * C3 + c] = (_Float16)mv;
    }
}

// ---------------------------------------------------------------------------
// Generic GEMM + BN + relu, all-f16 operands, f32 accumulate.
// Yh[M x COUT] = relu( Xh[M x KP] * Wh^T + bias2 ); KP multiple of 32,
// zero-padded; M multiple of 16. grid = (ceil(M/64), COUT/16), block 128.
// ---------------------------------------------------------------------------
__global__ __launch_bounds__(128) void gemm_f16_kernel(
    const _Float16* Xh, int M, int KP,
    const _Float16* Wh, const float* bias2,
    _Float16* Yh, int COUT)
{
    const int wave = threadIdx.x >> 5;
    const int lane = threadIdx.x & 31;
    const int gq   = lane >> 4;
    const int ln   = lane & 15;
    const int rowbase = blockIdx.x * 64 + wave * 16;
    if (rowbase >= M) return;                       // wave-uniform; EXEC stays full
    const int n = blockIdx.y * 16 + ln;

    const _Float16* xr = Xh + (size_t)(rowbase + ln) * KP;
    const _Float16* wr = Wh + (size_t)n * KP;
    const float b2 = bias2[n];

    v8f acc = {0.f,0.f,0.f,0.f,0.f,0.f,0.f,0.f};
    for (int k0 = 0; k0 < KP; k0 += 32) {
        v8h alo = *(const v8h*)(xr + k0 + 8 * gq);
        v8h ahi = *(const v8h*)(xr + k0 + 16 + 8 * gq);
        v8h blo = *(const v8h*)(wr + k0 + 8 * gq);
        v8h bhi = *(const v8h*)(wr + k0 + 16 + 8 * gq);
        v16h a, bf;
#pragma unroll
        for (int i = 0; i < 8; i++) {
            a[i] = alo[i];  a[i + 8]  = ahi[i];
            bf[i] = blo[i]; bf[i + 8] = bhi[i];
        }
        acc = __builtin_amdgcn_wmma_f32_16x16x32_f16(
            false, a, false, bf, (short)0, acc, false, false);
    }
#pragma unroll
    for (int r = 0; r < 8; r++) {
        int row = rowbase + r + 8 * gq;
        Yh[(size_t)row * COUT + n] = (_Float16)fmaxf(acc[r] + b2, 0.f);
    }
}

// ---------------------------------------------------------------------------
// Support kernels
// ---------------------------------------------------------------------------
__global__ void transpose_xyz_kernel(const float* xyz_in, float* l0_xyz)
{
    int t = blockIdx.x * blockDim.x + threadIdx.x;
    if (t >= BATCH * NPTS) return;
    int b = t / NPTS, n = t % NPTS;
#pragma unroll
    for (int d = 0; d < 3; d++)
        l0_xyz[(size_t)t * 3 + d] = xyz_in[((size_t)b * 3 + d) * NPTS + n];
}

__global__ __launch_bounds__(256) void fps_kernel(const float* xyz, int Np, int npoint, int* out)
{
    const int b   = blockIdx.x;
    const int tid = threadIdx.x;
    const float* px = xyz + (size_t)b * Np * 3;
    __shared__ float dist[4096];
    __shared__ float rv[256];
    __shared__ int   ri[256];
    __shared__ int   sfar;

    for (int i = tid; i < Np; i += 256) dist[i] = 1e10f;
    if (tid == 0) sfar = 0;
    __syncthreads();

    for (int j = 0; j < npoint; j++) {
        int far = sfar;
        if (tid == 0) out[b * npoint + j] = far;   // record BEFORE update (matches scan)
        float cx = px[far * 3], cy = px[far * 3 + 1], cz = px[far * 3 + 2];
        float bv = -1.f; int bi = 0;
        for (int i = tid; i < Np; i += 256) {
            float dx = px[i * 3] - cx, dy = px[i * 3 + 1] - cy, dz = px[i * 3 + 2] - cz;
            float d  = dx * dx + dy * dy + dz * dz;
            float nd = fminf(dist[i], d);
            dist[i] = nd;
            if (nd > bv) { bv = nd; bi = i; }
        }
        rv[tid] = bv; ri[tid] = bi;
        __syncthreads();
        for (int s = 128; s > 0; s >>= 1) {
            if (tid < s) {
                if (rv[tid + s] > rv[tid] ||
                    (rv[tid + s] == rv[tid] && ri[tid + s] < ri[tid])) {
                    rv[tid] = rv[tid + s]; ri[tid] = ri[tid + s];
                }
            }
            __syncthreads();
        }
        if (tid == 0) sfar = ri[0];
        __syncthreads();
    }
}

__global__ void gather_xyz_kernel(const float* xyz, const int* idx, float* out, int Np, int M)
{
    int t = blockIdx.x * blockDim.x + threadIdx.x;
    if (t >= BATCH * M) return;
    int b = t / M;
    int gi = idx[t];
#pragma unroll
    for (int d = 0; d < 3; d++)
        out[(size_t)t * 3 + d] = xyz[((size_t)b * Np + gi) * 3 + d];
}

__global__ void query_ball_kernel(const float* xyz, const float* new_xyz,
                                  int Np, int M, float r2, int ns, int* gidx)
{
    int t = blockIdx.x * blockDim.x + threadIdx.x;
    if (t >= BATCH * M) return;
    int b = t / M;
    const float* px = xyz + (size_t)b * Np * 3;
    float cx = new_xyz[t * 3], cy = new_xyz[t * 3 + 1], cz = new_xyz[t * 3 + 2];
    int* g = gidx + (size_t)t * ns;
    int cnt = 0, first = -1;
    for (int n = 0; n < Np && cnt < ns; n++) {
        float dx = px[n * 3] - cx, dy = px[n * 3 + 1] - cy, dz = px[n * 3 + 2] - cz;
        float d = dx * dx + dy * dy + dz * dz;
        if (d <= r2) { if (first < 0) first = n; g[cnt++] = n; }
    }
    if (first < 0) first = 0;
    for (int k = cnt; k < ns; k++) g[k] = first;
}

__global__ void knn3_kernel(const float* xyz1, const float* xyz2, int N1, int S2,
                            int* idx3, float* w3)
{
    int t = blockIdx.x * blockDim.x + threadIdx.x;
    if (t >= BATCH * N1) return;
    int b = t / N1;
    float cx = xyz1[t * 3], cy = xyz1[t * 3 + 1], cz = xyz1[t * 3 + 2];
    const float* p2 = xyz2 + (size_t)b * S2 * 3;
    float d0 = 1e30f, d1 = 1e30f, d2 = 1e30f;
    int   i0 = 0, i1 = 0, i2 = 0;
    for (int s = 0; s < S2; s++) {
        float dx = p2[s * 3] - cx, dy = p2[s * 3 + 1] - cy, dz = p2[s * 3 + 2] - cz;
        float d = dx * dx + dy * dy + dz * dz;
        if (d < d0)      { d2 = d1; i2 = i1; d1 = d0; i1 = i0; d0 = d; i0 = s; }
        else if (d < d1) { d2 = d1; i2 = i1; d1 = d; i1 = s; }
        else if (d < d2) { d2 = d; i2 = s; }
    }
    float r0 = 1.f / (d0 + 1e-8f), r1 = 1.f / (d1 + 1e-8f), r2f = 1.f / (d2 + 1e-8f);
    float inv = 1.f / (r0 + r1 + r2f);
    idx3[t * 3 + 0] = i0; idx3[t * 3 + 1] = i1; idx3[t * 3 + 2] = i2;
    w3[t * 3 + 0] = r0 * inv; w3[t * 3 + 1] = r1 * inv; w3[t * 3 + 2] = r2f * inv;
}

// concat kernels write f16 X buffers with zero-padded K-stride
__global__ void sa3_concat_kernel(const float* l2_xyz, const _Float16* l2p, _Float16* X)
{
    long t = (long)blockIdx.x * blockDim.x + threadIdx.x;
    long total = (long)BATCH * 128 * 288;
    if (t >= total) return;
    int row = (int)(t / 288), c = (int)(t % 288);
    _Float16 v = (_Float16)0.f;
    if (c < 3)        v = (_Float16)l2_xyz[row * 3 + c];
    else if (c < 259) v = l2p[(size_t)row * 256 + (c - 3)];
    X[t] = v;
}

__global__ void fp3_concat_kernel(const _Float16* l2p, const _Float16* l3p, _Float16* X)
{
    long t = (long)blockIdx.x * blockDim.x + threadIdx.x;
    long total = (long)BATCH * 128 * 1280;
    if (t >= total) return;
    int row = (int)(t / 1280), c = (int)(t % 1280);
    int b = row / 128;
    X[t] = (c < 256) ? l2p[(size_t)row * 256 + c] : l3p[(size_t)b * 1024 + (c - 256)];
}

__global__ void fp2_concat_kernel(const _Float16* l1p, const _Float16* l2pfp,
                                  const int* idx3, const float* w3, _Float16* X)
{
    long t = (long)blockIdx.x * blockDim.x + threadIdx.x;
    long total = (long)BATCH * 512 * 384;
    if (t >= total) return;
    int row = (int)(t / 384), c = (int)(t % 384);
    int b = row / 512;
    float v;
    if (c < 128) v = (float)l1p[(size_t)row * 128 + c];
    else {
        int c2 = c - 128;
        v = 0.f;
#pragma unroll
        for (int k = 0; k < 3; k++)
            v += w3[row * 3 + k] *
                 (float)l2pfp[((size_t)b * 128 + idx3[row * 3 + k]) * 256 + c2];
    }
    X[t] = (_Float16)v;
}

__global__ void fp1_concat_kernel(const float* cls, const float* l0_xyz,
                                  const _Float16* l1pfp, const int* idx3,
                                  const float* w3, _Float16* X)
{
    long t = (long)blockIdx.x * blockDim.x + threadIdx.x;
    long total = (long)BATCH * NPTS * 160;
    if (t >= total) return;
    int row = (int)(t / 160), c = (int)(t % 160);
    int b = row / NPTS;
    float v = 0.f;
    if      (c < 16) v = cls[b * 16 + c];
    else if (c < 19) v = l0_xyz[(size_t)row * 3 + (c - 16)];
    else if (c < 22) v = l0_xyz[(size_t)row * 3 + (c - 19)];
    else if (c < 150) {
        int c2 = c - 22;
#pragma unroll
        for (int k = 0; k < 3; k++)
            v += w3[row * 3 + k] *
                 (float)l1pfp[((size_t)b * 512 + idx3[row * 3 + k]) * 128 + c2];
    }
    X[t] = (_Float16)v;
}

__global__ void maxpool_bc_kernel(const _Float16* in, _Float16* out, int S, int C)
{
    int t = blockIdx.x * blockDim.x + threadIdx.x;
    if (t >= BATCH * C) return;
    int b = t / C, c = t % C;
    float mv = -1e30f;
    for (int s = 0; s < S; s++) mv = fmaxf(mv, (float)in[((size_t)b * S + s) * C + c]);
    out[t] = (_Float16)mv;
}

__global__ void gfc3_kernel(const _Float16* g2, const float* W, const float* bias, float* out)
{
    int t = blockIdx.x * blockDim.x + threadIdx.x;
    if (t >= BATCH) return;
    float acc = 0.f;
    for (int c = 0; c < 256; c++) acc += (float)g2[t * 256 + c] * W[c];
    out[t] = acc + bias[0];
}

__global__ void conv2_lsm_kernel(const _Float16* feat, const float* W, const float* bias,
                                 float* out)
{
    int t = blockIdx.x * blockDim.x + threadIdx.x;
    if (t >= BATCH * NPTS) return;
    const _Float16* f = feat + (size_t)t * 128;
    float logit[4];
#pragma unroll
    for (int o = 0; o < 4; o++) {
        float acc = bias[o];
        const float* w = W + o * 128;
        for (int c = 0; c < 128; c++) acc += (float)f[c] * w[c];
        logit[o] = acc;
    }
    float m = fmaxf(fmaxf(logit[0], logit[1]), fmaxf(logit[2], logit[3]));
    float s = 0.f;
#pragma unroll
    for (int o = 0; o < 4; o++) s += expf(logit[o] - m);
    float lse = logf(s);
#pragma unroll
    for (int o = 0; o < 4; o++) out[(size_t)t * 4 + o] = logit[o] - m - lse;
}

// ---------------------------------------------------------------------------
extern "C" void kernel_launch(void* const* d_in, const int* in_sizes, int n_in,
                              void* d_out, int out_size, void* d_ws, size_t ws_size,
                              hipStream_t stream)
{
    (void)in_sizes; (void)n_in; (void)out_size; (void)ws_size;
    const float* xyz_in = (const float*)d_in[0];
    const float* cls    = (const float*)d_in[1];
    auto P = [&](int i) { return (const float*)d_in[i]; };

    float* ws = (float*)d_ws;
    size_t off = 0;  // in floats; keep 16B alignment
    auto allocf = [&](size_t n) { float* p = ws + off; off += (n + 3) & ~(size_t)3; return p; };
    auto alloci = [&](size_t n) { int* p = (int*)(ws + off); off += (n + 3) & ~(size_t)3; return p; };
    auto alloch = [&](size_t n) { _Float16* p = (_Float16*)(ws + off); off += ((n + 1) / 2 + 3) & ~(size_t)3; return p; };

    // f32 geometry
    float* l0_xyz = allocf((size_t)BATCH * NPTS * 3);
    float* l1_xyz = allocf((size_t)BATCH * 512 * 3);
    float* l2_xyz = allocf((size_t)BATCH * 128 * 3);
    float* fp2_w  = allocf((size_t)BATCH * 512 * 3);
    float* fp1_w  = allocf((size_t)BATCH * NPTS * 3);
    // int index buffers
    int* fps1    = alloci((size_t)BATCH * 512);
    int* fps2    = alloci((size_t)BATCH * 128);
    int* gidx1   = alloci((size_t)BATCH * 512 * 32);
    int* gidx2   = alloci((size_t)BATCH * 128 * 64);
    int* fp2_idx = alloci((size_t)BATCH * 512 * 3);
    int* fp1_idx = alloci((size_t)BATCH * NPTS * 3);
    // f16 activations
    _Float16* l1_points = alloch((size_t)BATCH * 512 * 128);
    _Float16* l2_points = alloch((size_t)BATCH * 128 * 256);
    _Float16* l3_points = alloch((size_t)BATCH * 1024);
    _Float16* l2pfp     = alloch((size_t)BATCH * 128 * 256);
    _Float16* l1pfp     = alloch((size_t)BATCH * 512 * 128);
    _Float16* gbuf1     = alloch((size_t)BATCH * 512);
    _Float16* gbuf2     = alloch((size_t)BATCH * 256);
    _Float16* Xbuf      = alloch((size_t)BATCH * NPTS * 160);  // >= all X needs
    _Float16* Ybuf1     = alloch((size_t)BATCH * NPTS * 128);
    _Float16* Ybuf2     = alloch((size_t)BATCH * NPTS * 128);

    // folded f16 weights + f32 bias2, one pair per BN layer
    struct Layer { int base, K, KP, N; _Float16* Wh; float* b2; };
    Layer L[18] = {
        {2,   6,   32,   64, 0, 0}, {6,  64,  64,  64, 0, 0}, {10, 64,  64, 128, 0, 0},   // sa1
        {14,  131, 160, 128, 0, 0}, {18, 128, 128, 128, 0, 0}, {22, 128, 128, 256, 0, 0}, // sa2
        {26,  259, 288, 256, 0, 0}, {30, 256, 256, 512, 0, 0}, {34, 512, 512, 1024, 0, 0},// sa3
        {72, 1024, 1024, 512, 0, 0}, {76, 512, 512, 256, 0, 0},                           // gfc1/2
        {38, 1280, 1280, 256, 0, 0}, {42, 256, 256, 256, 0, 0},                           // fp3
        {46,  384, 384, 256, 0, 0}, {50, 256, 256, 128, 0, 0},                            // fp2
        {54,  150, 160, 128, 0, 0}, {58, 128, 128, 128, 0, 0}, {62, 128, 128, 128, 0, 0}, // fp1
    };
    for (int i = 0; i < 18; i++) { L[i].Wh = alloch((size_t)L[i].N * L[i].KP); L[i].b2 = allocf(L[i].N); }
    _Float16* headWh = alloch((size_t)128 * 128);
    float*    headB2 = allocf(128);

    float* out_lsm  = (float*)d_out;
    float* out_glob = out_lsm + (size_t)BATCH * NPTS * 4;

    auto GB = [](long total, int bs) { return (int)((total + bs - 1) / bs); };

    // ---- weight folding ----
    for (int i = 0; i < 18; i++) {
        long tot = (long)L[i].N * L[i].KP;
        prep_weights_kernel<<<GB(tot, 256), 256, 0, stream>>>(
            P(L[i].base), P(L[i].base + 1), P(L[i].base + 2), P(L[i].base + 3),
            L[i].K, L[i].KP, L[i].N, L[i].Wh, L[i].b2);
    }
    prep_weights_kernel<<<GB(128 * 128, 256), 256, 0, stream>>>(
        P(66), P(67), P(68), P(69), 128, 128, 128, headWh, headB2);

    auto gemm = [&](const _Float16* X, int M, int li, _Float16* Y) {
        dim3 grid(GB(M, 64), L[li].N / 16);
        gemm_f16_kernel<<<grid, 128, 0, stream>>>(X, M, L[li].KP, L[li].Wh, L[li].b2, Y, L[li].N);
    };

    // ---- l0 ----
    transpose_xyz_kernel<<<GB((long)BATCH * NPTS, 256), 256, 0, stream>>>(xyz_in, l0_xyz);

    // ---- SA1 ----
    fps_kernel<<<BATCH, 256, 0, stream>>>(l0_xyz, NPTS, 512, fps1);
    gather_xyz_kernel<<<GB((long)BATCH * 512, 256), 256, 0, stream>>>(l0_xyz, fps1, l1_xyz, NPTS, 512);
    query_ball_kernel<<<GB((long)BATCH * 512, 256), 256, 0, stream>>>(
        l0_xyz, l1_xyz, NPTS, 512, 0.2f * 0.2f, 32, gidx1);
    sa_fused_kernel<32, 6, 32, 64, 64, 128, 1><<<BATCH * 512, 128, 0, stream>>>(
        l0_xyz, l1_points, l1_xyz, gidx1,
        L[0].Wh, L[0].b2, L[1].Wh, L[1].b2, L[2].Wh, L[2].b2,
        l1_points, NPTS, 512);

    // ---- SA2 ----
    fps_kernel<<<BATCH, 256, 0, stream>>>(l1_xyz, 512, 128, fps2);
    gather_xyz_kernel<<<GB((long)BATCH * 128, 256), 256, 0, stream>>>(l1_xyz, fps2, l2_xyz, 512, 128);
    query_ball_kernel<<<GB((long)BATCH * 128, 256), 256, 0, stream>>>(
        l1_xyz, l2_xyz, 512, 128, 0.4f * 0.4f, 64, gidx2);
    sa_fused_kernel<64, 131, 160, 128, 128, 256, 2><<<BATCH * 128, 128, 0, stream>>>(
        l1_xyz, l1_points, l2_xyz, gidx2,
        L[3].Wh, L[3].b2, L[4].Wh, L[4].b2, L[5].Wh, L[5].b2,
        l2_points, 512, 128);

    // ---- SA3 (group-all) ----
    sa3_concat_kernel<<<GB((long)BATCH * 128 * 288, 256), 256, 0, stream>>>(l2_xyz, l2_points, Xbuf);
    gemm(Xbuf, BATCH * 128, 6, Ybuf1);   // 288 -> 256
    gemm(Ybuf1, BATCH * 128, 7, Ybuf2);  // 256 -> 512
    gemm(Ybuf2, BATCH * 128, 8, Xbuf);   // 512 -> 1024 (X no longer needed)
    maxpool_bc_kernel<<<GB((long)BATCH * 1024, 256), 256, 0, stream>>>(Xbuf, l3_points, 128, 1024);

    // ---- global head ----
    gemm(l3_points, BATCH, 9, gbuf1);    // 1024 -> 512
    gemm(gbuf1, BATCH, 10, gbuf2);       // 512 -> 256
    gfc3_kernel<<<1, 64, 0, stream>>>(gbuf2, P(80), P(81), out_glob);

    // ---- FP3 ----
    fp3_concat_kernel<<<GB((long)BATCH * 128 * 1280, 256), 256, 0, stream>>>(l2_points, l3_points, Xbuf);
    gemm(Xbuf, BATCH * 128, 11, Ybuf1);  // 1280 -> 256
    gemm(Ybuf1, BATCH * 128, 12, l2pfp); // 256 -> 256

    // ---- FP2 ----
    knn3_kernel<<<GB((long)BATCH * 512, 256), 256, 0, stream>>>(l1_xyz, l2_xyz, 512, 128, fp2_idx, fp2_w);
    fp2_concat_kernel<<<GB((long)BATCH * 512 * 384, 256), 256, 0, stream>>>(
        l1_points, l2pfp, fp2_idx, fp2_w, Xbuf);
    gemm(Xbuf, BATCH * 512, 13, Ybuf1);  // 384 -> 256
    gemm(Ybuf1, BATCH * 512, 14, l1pfp); // 256 -> 128

    // ---- FP1 ----
    knn3_kernel<<<GB((long)BATCH * NPTS, 256), 256, 0, stream>>>(l0_xyz, l1_xyz, NPTS, 512, fp1_idx, fp1_w);
    fp1_concat_kernel<<<GB((long)BATCH * NPTS * 160, 256), 256, 0, stream>>>(
        cls, l0_xyz, l1pfp, fp1_idx, fp1_w, Xbuf);
    gemm(Xbuf, BATCH * NPTS, 15, Ybuf1); // 160 -> 128
    gemm(Ybuf1, BATCH * NPTS, 16, Ybuf2);
    gemm(Ybuf2, BATCH * NPTS, 17, Xbuf); // l0_feat (X consumed)

    // ---- head + logits + log_softmax ----
    {
        dim3 grid(GB(BATCH * NPTS, 64), 128 / 16);
        gemm_f16_kernel<<<grid, 128, 0, stream>>>(Xbuf, BATCH * NPTS, 128, headWh, headB2, Ybuf1, 128);
    }
    conv2_lsm_kernel<<<GB((long)BATCH * NPTS, 256), 256, 0, stream>>>(Ybuf1, P(70), P(71), out_lsm);
}
